// LSTM_83751862272477
// MI455X (gfx1250) — matrix operand order repaired and checked by reading, not compiled, not run on hardware
//
#include <hip/hip_runtime.h>

// ---------------------------------------------------------------------------
// LSTM on MI455X (gfx1250): bf16 WMMA for both the input-projection GEMM and
// the per-step recurrent GEMM; f32 accumulate, f32 gates/cell state.
// Async global->LDS (ASYNCcnt) staging with double buffering.
// ---------------------------------------------------------------------------

typedef __attribute__((ext_vector_type(16))) __bf16 v16bf;
typedef __attribute__((ext_vector_type(8)))  __bf16 v8bf;
typedef __attribute__((ext_vector_type(8)))  float  v8f;

// Exact pointee type expected by __builtin_amdgcn_global_load_async_to_lds_b128
// (from the compiler diagnostic): int __attribute__((vector_size(16))).
typedef int i32x4_vs __attribute__((__vector_size__(16)));
typedef __attribute__((address_space(1))) i32x4_vs as1_v4i;
typedef __attribute__((address_space(3))) i32x4_vs as3_v4i;

#define LSTM_B   32
#define LSTM_T   2048
#define LSTM_D   512
#define LSTM_4D  2048
#define REC_NWG  32      // persistent workgroups in recurrence
#define REC_HPW  16      // hidden units per workgroup
#define REC_ROWS 64      // 4 gates * REC_HPW W_hh rows per workgroup

#if defined(__HIP_DEVICE_COMPILE__) && defined(__has_builtin)
#  if __has_builtin(__builtin_amdgcn_global_load_async_to_lds_b128) && \
      __has_builtin(__builtin_amdgcn_s_wait_asynccnt)
#    define HAVE_ASYNC_LDS 1
#  endif
#endif

// 16-byte global->LDS copy. Async path writes LDS directly (ASYNCcnt);
// fallback is load-to-VGPR + ds_store (LOADcnt/DScnt), synced by barriers.
static __device__ inline void async_cp_b128(const __bf16* gsrc, __bf16* lds_dst) {
#if defined(HAVE_ASYNC_LDS)
  __builtin_amdgcn_global_load_async_to_lds_b128(
      (as1_v4i*)(uintptr_t)gsrc,
      (as3_v4i*)(uintptr_t)lds_dst,
      0, 0);
#else
  *(v8bf*)lds_dst = *(const v8bf*)gsrc;
#endif
}

static __device__ inline void async_wait0() {
#if defined(HAVE_ASYNC_LDS)
  __builtin_amdgcn_s_wait_asynccnt(0);
#endif
}

// ---- WMMA fragment loaders (v_wmma_f32_16x16x32_bf16) ---------------------
// A matrix 16(M) x 32(K), source row-major with row stride ldm (elements).
// ISA: lane l, m=l%16; elements 0..7 = K kb..kb+7, 8..15 = K kb+16..kb+23,
// kb = 8*(l/16).
static __device__ inline v16bf wmma_load_a(const __bf16* base, int ldm) {
  const int lane = threadIdx.x & 31;
  const int r    = lane & 15;
  const int kb   = (lane >> 4) << 3;
  const __bf16* p = base + (size_t)r * ldm + kb;
  v8bf lo = *(const v8bf*)(p);
  v8bf hi = *(const v8bf*)(p + 16);
  v16bf v;
#pragma unroll
  for (int i = 0; i < 8; ++i) { v[i] = lo[i]; v[i + 8] = hi[i]; }
  return v;
}

// B matrix 32(K) x 16(N), supplied as row-major N x K (weight rows are N).
// ISA: lane l, n=l%16; elements 0..7 = K kb..kb+7, 8..15 = K kb+16..kb+23, kb = 8*(l/16).
static __device__ inline v16bf wmma_load_b(const __bf16* base, int ldm) {
  const int lane = threadIdx.x & 31;
  const int n    = lane & 15;
  const int kb   = (lane >> 4) << 3;
  const __bf16* p = base + (size_t)n * ldm + kb;
  v8bf lo = *(const v8bf*)(p);
  v8bf hi = *(const v8bf*)(p + 16);
  v16bf v;
#pragma unroll
  for (int i = 0; i < 8; ++i) { v[i] = lo[i]; v[i + 8] = hi[i]; }
  return v;
}

// C/D f32 16x16: lane l -> n = l%16, rows m = 8*(l/16) + r (r = vgpr 0..7).
static __device__ inline void wmma_store_lds(float* dst, int ld, v8f acc) {
  const int lane = threadIdx.x & 31;
  const int n    = lane & 15;
  const int m0   = (lane >> 4) << 3;
#pragma unroll
  for (int r = 0; r < 8; ++r) dst[(m0 + r) * ld + n] = acc[r];
}

static __device__ inline float fast_sigmoid(float x) { return 1.f / (1.f + __expf(-x)); }
static __device__ inline float fast_tanh(float x)    { return 1.f - 2.f / (__expf(2.f * x) + 1.f); }

// ---- f32 -> bf16 conversion (grid-stride) ---------------------------------
__global__ void lstm_cvt_bf16(const float* __restrict__ src, __bf16* __restrict__ dst,
                              long long n) {
  long long i = (long long)blockIdx.x * blockDim.x + threadIdx.x;
  long long s = (long long)gridDim.x * blockDim.x;
  for (; i < n; i += s) dst[i] = (__bf16)src[i];
}

__global__ void lstm_bias_init(const float* __restrict__ b_ih, const float* __restrict__ b_hh,
                               float* __restrict__ bias, __bf16* __restrict__ Hglob,
                               int* __restrict__ bar) {
  int i = blockIdx.x * blockDim.x + threadIdx.x;
  if (i < LSTM_4D) bias[i] = b_ih[i] + b_hh[i];
  if (i < 2 * LSTM_B * LSTM_D) Hglob[i] = (__bf16)0.f;   // h0 = 0, both ping-pong buffers
  if (i == 0) *bar = 0;                                  // barrier counter reset (per launch)
}

// ---- Phase 1: xg[t][b][n] = x[b,t,:] @ W_ih[n,:] + bias[n] ---------------
// M = B*T = 65536 (m = b*T + t), N = 4D = 2048, K = D = 512.
// Block tile 64x128, 8 waves (2x4), wave tile 32x32 (4 accumulators).
// K chunk 32, double-buffered LDS, async global->LDS staging.
static __device__ inline void xg_stage(const __bf16* __restrict__ Xb,
                                       const __bf16* __restrict__ Wih,
                                       __bf16 (*Ash)[40], __bf16 (*Bsh)[40],
                                       int m_base, int n_base, int kk, int tid) {
  const int arow = tid >> 2;               // 0..63
  const int ac8  = (tid & 3) << 3;         // 0,8,16,24
  async_cp_b128(Xb + (size_t)(m_base + arow) * LSTM_D + kk + ac8, &Ash[arow][ac8]);
  const int brow = tid >> 2;               // B rows 0..63 and 64..127
  async_cp_b128(Wih + (size_t)(n_base + brow) * LSTM_D + kk + ac8, &Bsh[brow][ac8]);
  async_cp_b128(Wih + (size_t)(n_base + brow + 64) * LSTM_D + kk + ac8, &Bsh[brow + 64][ac8]);
}

__global__ __launch_bounds__(256) void lstm_xg_gemm(
    const __bf16* __restrict__ Xb,    // [B*T][512]
    const __bf16* __restrict__ Wih,   // [2048][512]
    const float*  __restrict__ bias,  // [2048]
    float*        __restrict__ xg)    // [T][B][4D]
{
  __shared__ __bf16 Ash[2][64][40];     // 80B row stride: 16B aligned, bank-spread
  __shared__ __bf16 Bsh[2][128][40];
  const int tid    = threadIdx.x;
  const int wave   = tid >> 5;
  const int m_base = (blockIdx.x & 1023) * 64;   // 1024 M-blocks
  const int n_base = (blockIdx.x >> 10) * 128;   // 16 N-blocks
  const int m_off  = (wave & 1) * 32;
  const int n_off  = (wave >> 1) * 32;

  v8f acc00 = {0.f,0.f,0.f,0.f,0.f,0.f,0.f,0.f};
  v8f acc01 = acc00, acc10 = acc00, acc11 = acc00;

  xg_stage(Xb, Wih, Ash[0], Bsh[0], m_base, n_base, 0, tid);

  const int NK = LSTM_D / 32;   // 16
  for (int kc = 0; kc < NK; ++kc) {
    const int buf = kc & 1;
    async_wait0();              // my stage-kc copies done (ASYNCcnt in-order)
    __syncthreads();            // everyone's copies visible; prev compute done
    if (kc + 1 < NK)
      xg_stage(Xb, Wih, Ash[buf ^ 1], Bsh[buf ^ 1], m_base, n_base, (kc + 1) * 32, tid);

    v16bf a0 = wmma_load_a(&Ash[buf][m_off][0], 40);
    v16bf a1 = wmma_load_a(&Ash[buf][m_off + 16][0], 40);
    v16bf b0 = wmma_load_b(&Bsh[buf][n_off][0], 40);
    v16bf b1 = wmma_load_b(&Bsh[buf][n_off + 16][0], 40);
    acc00 = __builtin_amdgcn_wmma_f32_16x16x32_bf16(false, a0, false, b0, (short)0, acc00, false, false);
    acc01 = __builtin_amdgcn_wmma_f32_16x16x32_bf16(false, a0, false, b1, (short)0, acc01, false, false);
    acc10 = __builtin_amdgcn_wmma_f32_16x16x32_bf16(false, a1, false, b0, (short)0, acc10, false, false);
    acc11 = __builtin_amdgcn_wmma_f32_16x16x32_bf16(false, a1, false, b1, (short)0, acc11, false, false);
  }

  const int lane = tid & 31;
  const int n_ln = lane & 15;
  const int m0   = (lane >> 4) << 3;
  const int n0   = n_base + n_off + n_ln;
  const int n1   = n0 + 16;
  const float bi0 = bias[n0], bi1 = bias[n1];
#pragma unroll
  for (int r = 0; r < 8; ++r) {
    int mg0 = m_base + m_off + m0 + r;          // rows of a0
    int mg1 = mg0 + 16;                         // rows of a1
    size_t o0 = ((size_t)(mg0 & (LSTM_T - 1)) * LSTM_B + (mg0 >> 11)) * LSTM_4D;
    size_t o1 = ((size_t)(mg1 & (LSTM_T - 1)) * LSTM_B + (mg1 >> 11)) * LSTM_4D;
    xg[o0 + n0] = acc00[r] + bi0;               // 16 consecutive floats per lane group
    xg[o0 + n1] = acc01[r] + bi1;
    xg[o1 + n0] = acc10[r] + bi0;
    xg[o1 + n1] = acc11[r] + bi1;
  }
}

// ---- Phase 2: persistent recurrence -------------------------------------
// 32 workgroups; wg owns hidden units d in [wg*16, wg*16+16), i.e. W_hh rows
// {g*512 + wg*16 + j}, staged once into LDS (~65 KB of the WGP's 320 KB).
// Per step: GEMM h @ Wslice^T (16 WMMA/wave), gates, cell update (c in regs),
// h exchanged via ping-pong global buffer, device-wide atomic barrier.
__global__ __launch_bounds__(256) void lstm_recurrence(
    const float*  __restrict__ xg,     // [T][B][4D]
    const __bf16* __restrict__ Whh,    // [4D][D]
    __bf16*       __restrict__ Hglob,  // [2][B][D] ping-pong
    int*          __restrict__ bar,
    float*        __restrict__ out)    // [B][T][D]
{
  __shared__ __bf16 Wsl[REC_ROWS][520];   // 65 KB  (1040B stride: aligned, bank-spread)
  __shared__ __bf16 Hsh[LSTM_B][520];     // 32.5 KB
  __shared__ float  Gsh[LSTM_B][68];      // 8.5 KB gate pre-activations

  const int tid  = threadIdx.x;
  const int wg   = blockIdx.x;
  const int wave = tid >> 5;

  // Stage W_hh slice: 64 rows x 512 cols bf16 (one-time).
  for (int idx = tid; idx < REC_ROWS * (LSTM_D / 8); idx += 256) {
    int row = idx >> 6;                  // 0..63 local gate row (g*16 + j)
    int c8  = (idx & 63) << 3;
    int g = row >> 4, j = row & 15;
    const __bf16* src = Whh + (size_t)(g * LSTM_D + wg * REC_HPW + j) * LSTM_D + c8;
    *(v8bf*)&Wsl[row][c8] = *(const v8bf*)src;
  }
  float creg[2] = {0.f, 0.f};            // each thread owns 2 fixed (b, j) cells
  __syncthreads();

  const int m_off = (wave & 1) * 16;     // 2 M-tiles x 4 N-tiles over 8 waves
  const int n_off = (wave >> 1) * 16;

  for (int t = 0; t < LSTM_T; ++t) {
    // h_{t-1} from ping-pong buffer (t&1) into LDS (async global->LDS)
    const __bf16* Hin = Hglob + (size_t)(t & 1) * (LSTM_B * LSTM_D);
    for (int idx = tid; idx < LSTM_B * (LSTM_D / 8); idx += 256) {
      int row = idx >> 6;
      int c8  = (idx & 63) << 3;
      async_cp_b128(Hin + (size_t)row * LSTM_D + c8, &Hsh[row][c8]);
    }
    async_wait0();
    __syncthreads();

    v8f acc = {0.f,0.f,0.f,0.f,0.f,0.f,0.f,0.f};
#pragma unroll 4
    for (int kk = 0; kk < LSTM_D; kk += 32) {
      v16bf a = wmma_load_a(&Hsh[m_off][kk], 520);
      v16bf b = wmma_load_b(&Wsl[n_off][kk], 520);
      acc = __builtin_amdgcn_wmma_f32_16x16x32_bf16(false, a, false, b, (short)0, acc, false, false);
    }
    wmma_store_lds(&Gsh[m_off][n_off], 68, acc);
    __syncthreads();

    // Gates + state update: 512 (b, j) elements, 2 per thread.
    __bf16* Hout = Hglob + (size_t)((t + 1) & 1) * (LSTM_B * LSTM_D);
    const float* xgt = xg + (size_t)t * LSTM_B * LSTM_4D;
#pragma unroll
    for (int q = 0; q < 2; ++q) {
      int e = tid + q * 256;
      int b = e >> 4, j = e & 15;
      int d = wg * REC_HPW + j;
      const float* xgb = xgt + (size_t)b * LSTM_4D;
      float si = Gsh[b][ 0 + j] + xgb[0 * LSTM_D + d];
      float sf = Gsh[b][16 + j] + xgb[1 * LSTM_D + d];
      float sg = Gsh[b][32 + j] + xgb[2 * LSTM_D + d];
      float so = Gsh[b][48 + j] + xgb[3 * LSTM_D + d];
      float ig = fast_sigmoid(si);
      float fg = fast_sigmoid(sf);
      float gg = fast_tanh(sg);
      float og = fast_sigmoid(so);
      float c  = fg * creg[q] + ig * gg;
      creg[q]  = c;
      float h  = og * fast_tanh(c);
      out[((size_t)b * LSTM_T + t) * LSTM_D + d] = h;
      Hout[(size_t)b * LSTM_D + d] = (__bf16)h;
      // Prefetch next timestep's xg slice (global_prefetch_b8) to hide the
      // serial-loop L2/HBM latency.
      if (t + 1 < LSTM_T) {
        const float* nx = xgb + (size_t)LSTM_B * LSTM_4D;
        __builtin_prefetch(nx + 0 * LSTM_D + d, 0, 3);
        __builtin_prefetch(nx + 1 * LSTM_D + d, 0, 3);
        __builtin_prefetch(nx + 2 * LSTM_D + d, 0, 3);
        __builtin_prefetch(nx + 3 * LSTM_D + d, 0, 3);
      }
    }

    // Device-wide barrier (monotonic counter; reset each launch by init kernel).
    __threadfence();
    __syncthreads();
    if (tid == 0) {
      __hip_atomic_fetch_add(bar, 1, __ATOMIC_ACQ_REL, __HIP_MEMORY_SCOPE_AGENT);
      const int target = REC_NWG * (t + 1);
      while (__hip_atomic_load(bar, __ATOMIC_ACQUIRE, __HIP_MEMORY_SCOPE_AGENT) < target) {
        __builtin_amdgcn_s_sleep(1);
      }
    }
    __syncthreads();
  }
}

// ---------------------------------------------------------------------------
extern "C" void kernel_launch(void* const* d_in, const int* in_sizes, int n_in,
                              void* d_out, int out_size, void* d_ws, size_t ws_size,
                              hipStream_t stream) {
  const float* x    = (const float*)d_in[0];   // [B, T, D]
  const float* W_ih = (const float*)d_in[1];   // [4D, D]
  const float* W_hh = (const float*)d_in[2];   // [4D, D]
  const float* b_ih = (const float*)d_in[3];   // [4D]
  const float* b_hh = (const float*)d_in[4];   // [4D]
  float* out = (float*)d_out;                  // [B, T, D]

  char* ws = (char*)d_ws;
  size_t off = 0;
  float*  xg   = (float*)(ws + off);  off += (size_t)LSTM_T * LSTM_B * LSTM_4D * 4;  // 536 MB
  __bf16* Xb   = (__bf16*)(ws + off); off += (size_t)LSTM_B * LSTM_T * LSTM_D * 2;   // 67 MB
  __bf16* Wihb = (__bf16*)(ws + off); off += (size_t)LSTM_4D * LSTM_D * 2;           // 2 MB
  __bf16* Whhb = (__bf16*)(ws + off); off += (size_t)LSTM_4D * LSTM_D * 2;           // 2 MB
  float*  bias = (float*)(ws + off);  off += (size_t)LSTM_4D * 4;
  __bf16* Hg   = (__bf16*)(ws + off); off += (size_t)2 * LSTM_B * LSTM_D * 2;
  int*    bar  = (int*)(ws + off);    off += 256;

  const long long nx = (long long)LSTM_B * LSTM_T * LSTM_D;
  const long long nw = (long long)LSTM_4D * LSTM_D;

  lstm_cvt_bf16<<<2048, 256, 0, stream>>>(x, Xb, nx);
  lstm_cvt_bf16<<<512, 256, 0, stream>>>(W_ih, Wihb, nw);
  lstm_cvt_bf16<<<512, 256, 0, stream>>>(W_hh, Whhb, nw);
  lstm_bias_init<<<(2 * LSTM_B * LSTM_D + 255) / 256, 256, 0, stream>>>(b_ih, b_hh, bias, Hg, bar);
  lstm_xg_gemm<<<1024 * 16, 256, 0, stream>>>(Xb, Wihb, bias, xg);
  lstm_recurrence<<<REC_NWG, 256, 0, stream>>>(xg, Whhb, Hg, bar, out);
}
